// FSMN_9861244911960
// MI455X (gfx1250) — compile-verified, matrix operand order checked by reading, not run on hardware
//
#include <hip/hip_runtime.h>
#include <hip/hip_bf16.h>

// FSMN: out[b,t,o] = sum_d x[b,t,d]*W1[d,o] + sum_d hh[b,t,d]*W2[d,o] + bias[o]
// hh[b,t,d] = sum_{j<min(20,t+1)} mw[j]*x[b,t-j,d]
// Strategy: fuse as one bf16 WMMA GEMM  A'[32768,2048] @ W'[2048,1024]
// where A' = [x_bf16 | hh_bf16], W' = [W1;W2]_bf16, fp32 accumulate.
//
// Roofline (MI455X): 137.4 GFLOP matrix work -> compute-bound at dense-bf16
// WMMA rates (~55 us); min HBM traffic ~276 MiB (~12 us @ 23.3 TB/s).
// Grid is launched N-fastest so the 8 column-blocks sharing one A panel are
// co-resident and hit the 192 MB L2, keeping Apack HBM traffic near 1x.

typedef __attribute__((ext_vector_type(16))) __bf16          v16bf;
typedef __attribute__((ext_vector_type(16))) unsigned short  v16u;
typedef __attribute__((ext_vector_type(8)))  float           v8f;

#define Bdim   16
#define Tdim   2048
#define Ddim   1024
#define Odim   1024
#define Mtaps  20
#define KC     2048          // concatenated K
#define MTILES 2048          // 32768 rows / 16
#define KTILES 64            // 2048 / 32
#define NTILES 64            // 1024 / 16

// Per-lane K index inside a 16x32 bf16 A fragment (ISA 7.12.2, 16-bit A 16x32):
//   lanes 0-15:  v0..v3 hold K=0..7 (2/vgpr), v4..v7 hold K=16..23
//   lanes 16-31: v0..v3 hold K=8..15,         v4..v7 hold K=24..31
// The 32x16 B fragment uses the same K striping with N = lane%16 (mirrors 16-bit C/D).
__device__ __forceinline__ int frag_k(int lane, int i) {
    int v = i >> 1, h = i & 1;
    int base = (lane & 16) ? 8 : 0;
    return (v < 4) ? (base + 2 * v + h) : (16 + base + 2 * (v - 4) + h);
}

__device__ __forceinline__ unsigned short f32_to_bf16_rne(float f) {
    unsigned int u = __float_as_uint(f);
    u += 0x7fffu + ((u >> 16) & 1u);
    return (unsigned short)(u >> 16);
}

// One thread packs one 32-byte lane-fragment of A' (and computes the conv for the
// hh half on the fly).  gid -> (mtile, ktile, lane).
__global__ __launch_bounds__(256) void fsmn_pack_a(
    const float* __restrict__ x, const float* __restrict__ mw,
    unsigned short* __restrict__ Apack)
{
    int gid   = blockIdx.x * 256 + threadIdx.x;        // 2048*64*32 = 4,194,304 total
    int lane  = gid & 31;
    int ktile = (gid >> 5) & (KTILES - 1);
    int mtile = gid >> 11;
    int m     = (mtile << 4) + (lane & 15);
    int b     = m >> 11;                                // / Tdim
    int t     = m & (Tdim - 1);
    const float* xrow = x + ((size_t)b * Tdim + t) * Ddim;

    int jmax = (t + 1 < Mtaps) ? (t + 1) : Mtaps;

    v16u frag;
#pragma unroll
    for (int i = 0; i < 16; ++i) {
        int k = (ktile << 5) + frag_k(lane, i);
        float val;
        if (k < Ddim) {                                 // raw x half
            val = xrow[k];
        } else {                                        // causal-memory half
            int d = k - Ddim;
            float acc = 0.f;
            for (int j = 0; j < jmax; ++j)
                acc = fmaf(mw[j], xrow[d - j * Ddim], acc);
            val = acc;
        }
        frag[i] = f32_to_bf16_rne(val);
    }
    *(v16u*)(Apack + (size_t)gid * 16) = frag;          // 32B contiguous store
}

// Pack W' = [W1;W2] into B fragments.  gid -> (ktile, ntile, lane).
__global__ __launch_bounds__(256) void fsmn_pack_b(
    const float* __restrict__ W1, const float* __restrict__ W2,
    unsigned short* __restrict__ Bpack)
{
    int gid   = blockIdx.x * 256 + threadIdx.x;        // 64*64*32 = 131,072 total
    int lane  = gid & 31;
    int ntile = (gid >> 5) & (NTILES - 1);
    int ktile = gid >> 11;
    int n     = (ntile << 4) + (lane & 15);

    v16u frag;
#pragma unroll
    for (int i = 0; i < 16; ++i) {
        int k = (ktile << 5) + frag_k(lane, i);
        float val = (k < Ddim) ? W1[(size_t)k * Odim + n]
                               : W2[(size_t)(k - Ddim) * Odim + n];
        frag[i] = f32_to_bf16_rne(val);
    }
    *(v16u*)(Bpack + (size_t)gid * 16) = frag;
}

// GEMM: 8 waves/block in a 4(M) x 2(N) wave grid; each wave owns 4x4 fragments
// (64x64 fp32 tile).  Workgroup tile = 256(M) x 128(N).
// blockIdx.x = N-block (fast dispatch dim, 8 wide) -> blocks sharing an A panel
// are adjacent in dispatch order for L2 reuse; blockIdx.y = M-block (128).
__global__ __launch_bounds__(256) void fsmn_gemm(
    const unsigned short* __restrict__ Apack,
    const unsigned short* __restrict__ Bpack,
    const float* __restrict__ bias,
    float* __restrict__ out)
{
    const int lane   = threadIdx.x & 31;
    const int wave   = threadIdx.x >> 5;
    const int wm     = wave & 3;
    const int wn     = wave >> 2;
    const int mtile0 = blockIdx.y * 16 + wm * 4;
    const int ntile0 = blockIdx.x * 8  + wn * 4;

    const v16bf* __restrict__ Af = (const v16bf*)Apack;
    const v16bf* __restrict__ Bf = (const v16bf*)Bpack;

    v8f acc[4][4] = {};

    for (int kt = 0; kt < KTILES; ++kt) {
        v16bf a[4], bb[4];
#pragma unroll
        for (int i = 0; i < 4; ++i)
            a[i] = Af[((mtile0 + i) * KTILES + kt) * 32 + lane];
#pragma unroll
        for (int j = 0; j < 4; ++j)
            bb[j] = Bf[(kt * NTILES + (ntile0 + j)) * 32 + lane];
#pragma unroll
        for (int i = 0; i < 4; ++i)
#pragma unroll
            for (int j = 0; j < 4; ++j)
                acc[i][j] = __builtin_amdgcn_wmma_f32_16x16x32_bf16(
                    /*neg_a=*/false, a[i], /*neg_b=*/false, bb[j],
                    /*c_mod=*/(short)0, acc[i][j],
                    /*reuse_a=*/false, /*reuse_b=*/false);
    }

    // Epilogue: C/D fp32 layout -> lanes 0-15: M=vgpr, lanes 16-31: M=vgpr+8; N=lane%16
#pragma unroll
    for (int j = 0; j < 4; ++j) {
        int n = ((ntile0 + j) << 4) + (lane & 15);
        float bv = bias[n];
#pragma unroll
        for (int i = 0; i < 4; ++i) {
            int mbase = ((mtile0 + i) << 4) + ((lane & 16) >> 1);
            float* orow = out + (size_t)mbase * Odim + n;
#pragma unroll
            for (int r = 0; r < 8; ++r)
                orow[(size_t)r * Odim] = acc[i][j][r] + bv;
        }
    }
}

extern "C" void kernel_launch(void* const* d_in, const int* in_sizes, int n_in,
                              void* d_out, int out_size, void* d_ws, size_t ws_size,
                              hipStream_t stream) {
    const float* x    = (const float*)d_in[0];   // [16,2048,1024]
    const float* W1   = (const float*)d_in[1];   // [1024,1024]
    const float* W2   = (const float*)d_in[2];   // [1024,1024]
    const float* bias = (const float*)d_in[3];   // [1024]
    const float* mw   = (const float*)d_in[4];   // [20]

    unsigned short* Apack = (unsigned short*)d_ws;                    // 128 MiB
    unsigned short* Bpack = Apack + (size_t)MTILES * KTILES * 512;    // + 4 MiB

    fsmn_pack_a<<<(MTILES * KTILES * 32) / 256, 256, 0, stream>>>(x, mw, Apack);
    fsmn_pack_b<<<(KTILES * NTILES * 32) / 256, 256, 0, stream>>>(W1, W2, Bpack);

    dim3 grid(NTILES / 8, MTILES / 16);   // (8, 128): N fastest for A-panel L2 reuse
    fsmn_gemm<<<grid, 256, 0, stream>>>(Apack, Bpack, bias, (float*)d_out);
}
